// MoELayer_13039520710827
// MI455X (gfx1250) — matrix operand order repaired and checked by reading, compiled, not run on hardware
//
#include <hip/hip_runtime.h>
#include <hip/hip_bf16.h>
#include <math.h>

// Problem constants (DeepSeek-V3 style MoE, B=4 S=2048 D=1024 F=512 E=64)
#define T_TOK     8192
#define DIM       1024
#define FDIM      512
#define NEXP      64
#define TOPKN     8
#define NGROUP    8
#define TOPKG     4
#define GSIZE     (NEXP / NGROUP)       // 8 experts per group
#define ROWS      (T_TOK * TOPKN)       // 65536 dispatched rows
#define TILE_M    64
#define TILE_N    128
#define KCH       32                    // bf16 WMMA K per step
#define MAX_TILES (ROWS / TILE_M + NEXP)  // 1088 upper bound
#define ROUTED_SCALE 2.5f

typedef __bf16 v16bf __attribute__((ext_vector_type(16)));
typedef float  v8f   __attribute__((ext_vector_type(8)));

__device__ __forceinline__ unsigned pack2_bf16(float a, float b) {
    unsigned ua = __float_as_uint(a);
    ua += 0x7FFFu + ((ua >> 16) & 1u);          // RNE
    unsigned ub = __float_as_uint(b);
    ub += 0x7FFFu + ((ub >> 16) & 1u);
    return (ua >> 16) | (ub & 0xFFFF0000u);
}

__device__ __forceinline__ __bf16 to_bf16(float f) {
    unsigned u = __float_as_uint(f);
    u += 0x7FFFu + ((u >> 16) & 1u);
    unsigned short hs = (unsigned short)(u >> 16);
    __bf16 r;
    __builtin_memcpy(&r, &hs, sizeof(r));
    return r;
}

__device__ __forceinline__ float silu_f(float g) { return g / (1.0f + __expf(-g)); }

// WMMA swizzle: destination (lane, slot) inside a [32][16] bf16 fragment block
// A-matrix 16x32: lane = (m&15) + 16*((k>>3)&1); slot = (k&7) + 8*(k>>4)
// B-matrix 32x16: lane = (n&15) + 16*((k>>3)&1); slot = (k&7) + 8*(k>>4)
__device__ __forceinline__ int a_lane(int m, int k) { return (m & 15) + 16 * ((k >> 3) & 1); }
__device__ __forceinline__ int a_slot(int k)        { return (k & 7) + 8 * (k >> 4); }
__device__ __forceinline__ int b_lane(int n, int k) { return (n & 15) + 16 * ((k >> 3) & 1); }
__device__ __forceinline__ int b_slot(int k)        { return (k & 7) + 8 * (k >> 4); }

// ---------------------------------------------------------------- K0: zero
__global__ void zero_kernel(float* __restrict__ out, int* __restrict__ counts) {
    int i = blockIdx.x * blockDim.x + threadIdx.x;
    if (i < T_TOK * DIM) out[i] = 0.0f;
    if (i < NEXP) counts[i] = 0;
}

// ------------------------------------------------- K1: router scores (sigmoid)
__global__ __launch_bounds__(256)
void router_scores(const float* __restrict__ x, const float* __restrict__ wr,
                   float* __restrict__ scores) {
    __shared__ float xs[4][DIM];
    int t0 = blockIdx.x * 4;
    for (int i = threadIdx.x; i < 4 * DIM; i += 256)
        xs[i >> 10][i & (DIM - 1)] = x[(size_t)(t0 + (i >> 10)) * DIM + (i & (DIM - 1))];
    __syncthreads();
    int lt = threadIdx.x >> 6;
    int e  = threadIdx.x & 63;
    float acc = 0.0f;
    for (int d = 0; d < DIM; ++d) acc += xs[lt][d] * wr[(size_t)d * NEXP + e];
    scores[(size_t)(t0 + lt) * NEXP + e] = 1.0f / (1.0f + __expf(-acc));
}

// --------------------------------------- K2: group-limited top-k per token
__global__ void route_topk(const float* __restrict__ scores, const float* __restrict__ br,
                           int* __restrict__ topk_idx, float* __restrict__ topk_w,
                           int* __restrict__ counts) {
    int t = blockIdx.x * blockDim.x + threadIdx.x;
    if (t >= T_TOK) return;
    float s[NEXP], swb[NEXP];
    for (int e = 0; e < NEXP; ++e) {
        s[e] = scores[(size_t)t * NEXP + e];
        swb[e] = s[e] + br[e];
    }
    float gs[NGROUP];
    for (int g = 0; g < NGROUP; ++g) {
        float m1 = -1e30f, m2 = -1e30f;
        for (int j = 0; j < GSIZE; ++j) {
            float v = swb[g * GSIZE + j];
            if (v > m1) { m2 = m1; m1 = v; } else if (v > m2) { m2 = v; }
        }
        gs[g] = m1 + m2;
    }
    bool gsel[NGROUP];
    for (int g = 0; g < NGROUP; ++g) gsel[g] = false;
    for (int it = 0; it < TOPKG; ++it) {
        int bg = 0; float bv = -1e30f;
        for (int g = 0; g < NGROUP; ++g)
            if (!gsel[g] && gs[g] > bv) { bv = gs[g]; bg = g; }
        gsel[bg] = true;
    }
    float ms[NEXP];
    for (int e = 0; e < NEXP; ++e) ms[e] = gsel[e >> 3] ? swb[e] : 0.0f;
    int   ti[TOPKN];
    float tw[TOPKN];
    float wsum = 0.0f;
    for (int k = 0; k < TOPKN; ++k) {
        int be = 0; float bv = -1e30f;
        for (int e = 0; e < NEXP; ++e)
            if (ms[e] > bv) { bv = ms[e]; be = e; }
        ms[be] = -1e30f;
        ti[k] = be;
        tw[k] = s[be];
        wsum += tw[k];
    }
    for (int k = 0; k < TOPKN; ++k) {
        topk_idx[(size_t)t * TOPKN + k] = ti[k];
        topk_w[(size_t)t * TOPKN + k]   = ROUTED_SCALE * tw[k] / (wsum + 1e-20f);
        atomicAdd(&counts[ti[k]], 1);
    }
}

// ------------------------------ K3: prefix sum + tile table (single thread)
__global__ void build_tiles(const int* __restrict__ counts, int* __restrict__ offsets,
                            int* __restrict__ cursor, int* __restrict__ tile_tab,
                            int* __restrict__ n_tiles) {
    if (threadIdx.x != 0 || blockIdx.x != 0) return;
    int off = 0, nt = 0;
    for (int e = 0; e < NEXP; ++e) {
        offsets[e] = off;
        cursor[e]  = 0;
        int n = counts[e];
        for (int t0 = 0; t0 < n; t0 += TILE_M) {
            tile_tab[nt * 3 + 0] = e;
            tile_tab[nt * 3 + 1] = off + t0;
            tile_tab[nt * 3 + 2] = (n - t0 < TILE_M) ? (n - t0) : TILE_M;
            ++nt;
        }
        off += n;
    }
    *n_tiles = nt;
}

// -------------------------------------------------- K4: expert-sorted scatter
__global__ void scatter_kernel(const int* __restrict__ topk_idx, const float* __restrict__ topk_w,
                               const int* __restrict__ offsets, int* __restrict__ cursor,
                               int* __restrict__ row_tok, float* __restrict__ row_w) {
    int i = blockIdx.x * blockDim.x + threadIdx.x;
    if (i >= ROWS) return;
    int t = i >> 3;
    int e = topk_idx[i];
    int pos = offsets[e] + atomicAdd(&cursor[e], 1);
    row_tok[pos] = t;
    row_w[pos]   = topk_w[i];
}

// --------------------- K5: gate+up grouped GEMM + fused SwiGLU (bf16 WMMA)
// Double-buffered, WMMA-swizzled LDS: fragments load as contiguous ds_load_b128.
__global__ __launch_bounds__(256)
void gate_up_kernel(const float* __restrict__ x, const float* __restrict__ wg,
                    const float* __restrict__ wu, const int* __restrict__ tile_tab,
                    const int* __restrict__ n_tiles, const int* __restrict__ row_tok,
                    __bf16* __restrict__ h, int shared_mode) {
    __shared__ __align__(16) __bf16 As[2][4][32][16];   // [buf][mb][lane][slot]
    __shared__ __align__(16) __bf16 Bgs[2][8][32][16];  // [buf][nb][lane][slot]
    __shared__ __align__(16) __bf16 Bus[2][8][32][16];
    __shared__ int tokl[TILE_M];
    const int ABUF = (int)sizeof(As[0]);   // 4096 B: buf stride
    const int BBUF = (int)sizeof(Bgs[0]);  // 8192 B

    int e, rs, nr;
    if (shared_mode) { e = 0; rs = blockIdx.y * TILE_M; nr = TILE_M; }
    else {
        if ((int)blockIdx.y >= *n_tiles) return;
        e  = tile_tab[blockIdx.y * 3 + 0];
        rs = tile_tab[blockIdx.y * 3 + 1];
        nr = tile_tab[blockIdx.y * 3 + 2];
    }
    const int f0 = blockIdx.x * TILE_N;
    const float* wgb = wg + (size_t)e * DIM * FDIM;
    const float* wub = wu + (size_t)e * DIM * FDIM;

    const int tid = threadIdx.x;
    if (tid < TILE_M) tokl[tid] = shared_mode ? (rs + tid) : row_tok[rs + tid];
    __syncthreads();

    // loop-invariant staging addresses (buf0 only; buf1 = +ABUF/+BBUF bytes)
    const float* arow[4];
    char* adst0[4];
    #pragma unroll
    for (int it = 0; it < 4; ++it) {
        int idx = tid + it * 256;
        int m  = idx >> 4;
        int kp = (idx & 15) * 2;
        arow[it]  = (m < nr) ? &x[(size_t)tokl[m] * DIM + kp] : nullptr;
        adst0[it] = (char*)&As[0][m >> 4][a_lane(m, kp)][a_slot(kp)];
    }
    const int bn = tid & 127;           // fixed column within tile
    char*  bgdst0[8];
    char*  budst0[8];
    size_t bsrc[8];
    #pragma unroll
    for (int it = 0; it < 8; ++it) {
        int k = ((tid >> 7) + it * 2) * 2;   // even k, 0..30
        bsrc[it]   = (size_t)k * FDIM + f0 + bn;
        bgdst0[it] = (char*)&Bgs[0][bn >> 4][b_lane(bn, k)][b_slot(k)];
        budst0[it] = (char*)&Bus[0][bn >> 4][b_lane(bn, k)][b_slot(k)];
    }

    auto stage = [&](int kc, int buf) {
        const int ao = buf * ABUF, bo = buf * BBUF;
        #pragma unroll
        for (int it = 0; it < 4; ++it) {
            float v0 = 0.0f, v1 = 0.0f;
            if (arow[it]) { v0 = arow[it][kc * KCH]; v1 = arow[it][kc * KCH + 1]; }
            *(unsigned*)(adst0[it] + ao) = pack2_bf16(v0, v1);
        }
        size_t ko = (size_t)kc * KCH * FDIM;
        #pragma unroll
        for (int it = 0; it < 8; ++it) {
            size_t g0 = ko + bsrc[it];
            *(unsigned*)(bgdst0[it] + bo) = pack2_bf16(wgb[g0], wgb[g0 + FDIM]);
            *(unsigned*)(budst0[it] + bo) = pack2_bf16(wub[g0], wub[g0 + FDIM]);
        }
    };

    const int lane = tid & 31;
    const int wave = tid >> 5;
    const int mb   = wave >> 1;     // 0..3: 16-row block
    const int nh   = wave & 1;      // 0..1: 64-col half
    const int lh   = lane >> 4;
    const int l16  = lane & 15;

    v8f accg[4], accu[4];
    #pragma unroll
    for (int i = 0; i < 4; ++i)
        #pragma unroll
        for (int r = 0; r < 8; ++r) { accg[i][r] = 0.0f; accu[i][r] = 0.0f; }

    const int NK = DIM / KCH;
    stage(0, 0);
    __syncthreads();
    for (int kc = 0; kc < NK; ++kc) {
        int cur = kc & 1;
        if (kc + 1 < NK) {
            stage(kc + 1, cur ^ 1);
            if (kc + 2 < NK) {
                __builtin_prefetch(&wgb[(size_t)(kc + 2) * KCH * FDIM + bsrc[0]], 0, 3);
                __builtin_prefetch(&wub[(size_t)(kc + 2) * KCH * FDIM + bsrc[0]], 0, 3);
            }
        }
        v16bf a = *reinterpret_cast<const v16bf*>(&As[cur][mb][lane][0]);
        #pragma unroll
        for (int nb = 0; nb < 4; ++nb) {
            v16bf bg = *reinterpret_cast<const v16bf*>(&Bgs[cur][nh * 4 + nb][lane][0]);
            v16bf bu = *reinterpret_cast<const v16bf*>(&Bus[cur][nh * 4 + nb][lane][0]);
            accg[nb] = __builtin_amdgcn_wmma_f32_16x16x32_bf16(
                false, a, false, bg, (short)0, accg[nb], false, false);
            accu[nb] = __builtin_amdgcn_wmma_f32_16x16x32_bf16(
                false, a, false, bu, (short)0, accu[nb], false, false);
        }
        __syncthreads();
    }

    // fused SwiGLU epilogue -> bf16 intermediate h
    #pragma unroll
    for (int nb = 0; nb < 4; ++nb)
        #pragma unroll
        for (int r = 0; r < 8; ++r) {
            int row = mb * 16 + r + lh * 8;
            if (row < nr) {
                int col = f0 + nh * 64 + nb * 16 + l16;
                float g = accg[nb][r], u = accu[nb][r];
                h[(size_t)(rs + row) * FDIM + col] = to_bf16(silu_f(g) * u);
            }
        }
}

// -------------------- K6: down grouped GEMM + weighted scatter-add (bf16 WMMA)
__global__ __launch_bounds__(256)
void down_kernel(const __bf16* __restrict__ h, const float* __restrict__ wd,
                 const int* __restrict__ tile_tab, const int* __restrict__ n_tiles,
                 const int* __restrict__ row_tok, const float* __restrict__ row_w,
                 float* __restrict__ out, int shared_mode) {
    __shared__ __align__(16) __bf16 As[2][4][32][16];
    __shared__ __align__(16) __bf16 Bds[2][8][32][16];
    __shared__ int   tokl[TILE_M];
    __shared__ float wl[TILE_M];
    const int ABUF = (int)sizeof(As[0]);
    const int BBUF = (int)sizeof(Bds[0]);

    int e, rs, nr;
    if (shared_mode) { e = 0; rs = blockIdx.y * TILE_M; nr = TILE_M; }
    else {
        if ((int)blockIdx.y >= *n_tiles) return;
        e  = tile_tab[blockIdx.y * 3 + 0];
        rs = tile_tab[blockIdx.y * 3 + 1];
        nr = tile_tab[blockIdx.y * 3 + 2];
    }
    const int d0 = blockIdx.x * TILE_N;
    const float* wdb = wd + (size_t)e * FDIM * DIM;

    const int tid = threadIdx.x;
    if (tid < TILE_M) {
        tokl[tid] = shared_mode ? (rs + tid) : row_tok[rs + tid];
        wl[tid]   = shared_mode ? 1.0f : row_w[rs + tid];
    }
    __syncthreads();

    const __bf16* arow[4];
    char* adst0[4];
    #pragma unroll
    for (int it = 0; it < 4; ++it) {
        int idx = tid + it * 256;
        int m  = idx >> 4;
        int kp = (idx & 15) * 2;
        arow[it]  = (m < nr) ? &h[(size_t)(rs + m) * FDIM + kp] : nullptr;
        adst0[it] = (char*)&As[0][m >> 4][a_lane(m, kp)][a_slot(kp)];
    }
    const int bn = tid & 127;
    char*  bdst0[8];
    size_t bsrc[8];
    #pragma unroll
    for (int it = 0; it < 8; ++it) {
        int k = ((tid >> 7) + it * 2) * 2;
        bsrc[it]  = (size_t)k * DIM + d0 + bn;
        bdst0[it] = (char*)&Bds[0][bn >> 4][b_lane(bn, k)][b_slot(k)];
    }

    auto stage = [&](int kc, int buf) {
        const int ao = buf * ABUF, bo = buf * BBUF;
        #pragma unroll
        for (int it = 0; it < 4; ++it) {
            unsigned v = 0u;
            if (arow[it]) v = *(const unsigned*)&arow[it][kc * KCH];  // bf16 pair copy
            *(unsigned*)(adst0[it] + ao) = v;
        }
        size_t ko = (size_t)kc * KCH * DIM;
        #pragma unroll
        for (int it = 0; it < 8; ++it) {
            size_t g0 = ko + bsrc[it];
            *(unsigned*)(bdst0[it] + bo) = pack2_bf16(wdb[g0], wdb[g0 + DIM]);
        }
    };

    const int lane = tid & 31;
    const int wave = tid >> 5;
    const int mb   = wave >> 1;
    const int nh   = wave & 1;
    const int lh   = lane >> 4;
    const int l16  = lane & 15;

    v8f acc[4];
    #pragma unroll
    for (int i = 0; i < 4; ++i)
        #pragma unroll
        for (int r = 0; r < 8; ++r) acc[i][r] = 0.0f;

    const int NK = FDIM / KCH;
    stage(0, 0);
    __syncthreads();
    for (int kc = 0; kc < NK; ++kc) {
        int cur = kc & 1;
        if (kc + 1 < NK) {
            stage(kc + 1, cur ^ 1);
            if (kc + 2 < NK)
                __builtin_prefetch(&wdb[(size_t)(kc + 2) * KCH * DIM + bsrc[0]], 0, 3);
        }
        v16bf a = *reinterpret_cast<const v16bf*>(&As[cur][mb][lane][0]);
        #pragma unroll
        for (int nb = 0; nb < 4; ++nb) {
            v16bf bf = *reinterpret_cast<const v16bf*>(&Bds[cur][nh * 4 + nb][lane][0]);
            acc[nb] = __builtin_amdgcn_wmma_f32_16x16x32_bf16(
                false, a, false, bf, (short)0, acc[nb], false, false);
        }
        __syncthreads();
    }

    #pragma unroll
    for (int nb = 0; nb < 4; ++nb)
        #pragma unroll
        for (int r = 0; r < 8; ++r) {
            int row = mb * 16 + r + lh * 8;
            if (row < nr) {
                int col = d0 + nh * 64 + nb * 16 + l16;
                float v = acc[nb][r] * wl[row];
                atomicAdd(&out[(size_t)tokl[row] * DIM + col], v);
            }
        }
}

// ----------------------------------------------------------------- launcher
extern "C" void kernel_launch(void* const* d_in, const int* in_sizes, int n_in,
                              void* d_out, int out_size, void* d_ws, size_t ws_size,
                              hipStream_t stream) {
    const float* x   = (const float*)d_in[0];
    const float* wr  = (const float*)d_in[1];
    const float* br  = (const float*)d_in[2];
    const float* weg = (const float*)d_in[3];
    const float* weu = (const float*)d_in[4];
    const float* wed = (const float*)d_in[5];
    const float* wsg = (const float*)d_in[6];
    const float* wsu = (const float*)d_in[7];
    const float* wsd = (const float*)d_in[8];
    float* out = (float*)d_out;

    char* ws = (char*)d_ws;
    size_t off = 0;
    auto alloc = [&](size_t bytes) -> void* {
        void* p = ws + off;
        off = (off + bytes + 255) & ~(size_t)255;
        return p;
    };
    float*  scores   = (float*)alloc(sizeof(float) * (size_t)T_TOK * NEXP);
    int*    topk_idx = (int*)  alloc(sizeof(int)   * (size_t)T_TOK * TOPKN);
    float*  topk_w   = (float*)alloc(sizeof(float) * (size_t)T_TOK * TOPKN);
    int*    counts   = (int*)  alloc(sizeof(int) * NEXP);
    int*    offsets  = (int*)  alloc(sizeof(int) * NEXP);
    int*    cursor   = (int*)  alloc(sizeof(int) * NEXP);
    int*    n_tiles  = (int*)  alloc(sizeof(int));
    int*    tile_tab = (int*)  alloc(sizeof(int) * MAX_TILES * 3);
    int*    row_tok  = (int*)  alloc(sizeof(int)   * (size_t)ROWS);
    float*  row_w    = (float*)alloc(sizeof(float) * (size_t)ROWS);
    __bf16* h_rout   = (__bf16*)alloc(sizeof(__bf16) * (size_t)ROWS * FDIM);
    __bf16* h_sh     = (__bf16*)alloc(sizeof(__bf16) * (size_t)T_TOK * FDIM);

    zero_kernel<<<(T_TOK * DIM + 255) / 256, 256, 0, stream>>>(out, counts);
    router_scores<<<T_TOK / 4, 256, 0, stream>>>(x, wr, scores);
    route_topk<<<T_TOK / 256, 256, 0, stream>>>(scores, br, topk_idx, topk_w, counts);
    build_tiles<<<1, 1, 0, stream>>>(counts, offsets, cursor, tile_tab, n_tiles);
    scatter_kernel<<<ROWS / 256, 256, 0, stream>>>(topk_idx, topk_w, offsets, cursor, row_tok, row_w);
    // routed experts
    gate_up_kernel<<<dim3(FDIM / TILE_N, MAX_TILES), 256, 0, stream>>>(
        x, weg, weu, tile_tab, n_tiles, row_tok, h_rout, 0);
    down_kernel<<<dim3(DIM / TILE_N, MAX_TILES), 256, 0, stream>>>(
        h_rout, wed, tile_tab, n_tiles, row_tok, row_w, out, 0);
    // shared expert (identity rows, weight 1)
    gate_up_kernel<<<dim3(FDIM / TILE_N, T_TOK / TILE_M), 256, 0, stream>>>(
        x, wsg, wsu, tile_tab, n_tiles, nullptr, h_sh, 1);
    down_kernel<<<dim3(DIM / TILE_N, T_TOK / TILE_M), 256, 0, stream>>>(
        h_sh, wsd, tile_tab, n_tiles, nullptr, nullptr, out, 1);
}
